// MultiHeadAttention_79972291051868
// MI455X (gfx1250) — compile-verified, matrix operand order checked by reading, not run on hardware
//
#include <hip/hip_runtime.h>
#include <hip/hip_bf16.h>

// ---------------------------------------------------------------- types
typedef __bf16 bf16_t;
typedef __attribute__((ext_vector_type(16))) __bf16 bf16x16;
typedef __attribute__((ext_vector_type(8)))  __bf16 bf16x8;
typedef __attribute__((ext_vector_type(8)))  float  f32x8;
typedef __attribute__((ext_vector_type(4)))  float  f32x4;

#define BB   8
#define SS   1024
#define DD   1024
#define HH   16
#define HD   64
#define LDP  48      // padded LDS row stride (elements): 96B = 16B-aligned, 24-bank shift

// ---------------------------------------------------------------- helpers
__device__ __forceinline__ f32x8 wmma_bf16(bf16x16 a, bf16x16 b, f32x8 c) {
  return __builtin_amdgcn_wmma_f32_16x16x32_bf16(
      /*neg_a=*/false, a, /*neg_b=*/false, b,
      /*c_mod=*/(short)0, c, /*reuse_a=*/false, /*reuse_b=*/false);
}

// Async global -> LDS copy of 16B (CDNA5, ASYNCcnt-tracked).
__device__ __forceinline__ void async_copy_b128(unsigned lds_byte_off, const bf16_t* g) {
  asm volatile("global_load_async_to_lds_b128 %0, %1, off"
               :: "v"(lds_byte_off), "v"(g) : "memory");
}
__device__ __forceinline__ void wait_async_le(int n) {
  asm volatile("s_wait_asynccnt %0" :: "n"(n) : "memory");
}
// Low 32 bits of a flat-shared pointer == LDS byte offset (ISA 10.2 aperture map).
__device__ __forceinline__ unsigned lds_off(const void* p) {
  return (unsigned)(unsigned long long)p;
}

// Load one 16x32 bf16 A/B fragment from row-major global storage.
// 16-bit fragment layout (ISA 7.12.2): elems 0..7 -> K = half*8 + e,
// elems 8..15 -> K = 16 + half*8 + (e-8): two contiguous 16-byte runs.
__device__ __forceinline__ bf16x16 load_frag(const bf16_t* __restrict__ base,
                                             int row, int ld, int k0, int half) {
  const bf16_t* p = base + (size_t)row * ld + k0 + half * 8;
  bf16x8 lo = *(const bf16x8*)(p);
  bf16x8 hi = *(const bf16x8*)(p + 16);
  bf16x16 r;
#pragma unroll
  for (int i = 0; i < 8; ++i) { r[i] = lo[i]; r[i + 8] = hi[i]; }
  return r;
}

// Same fragment from an LDS tile with padded stride LDP (row length used = 32).
__device__ __forceinline__ bf16x16 lds_frag(const bf16_t* base, int row, int half) {
  const bf16_t* p = base + row * LDP + half * 8;
  bf16x8 lo = *(const bf16x8*)(p);
  bf16x8 hi = *(const bf16x8*)(p + 16);
  bf16x16 r;
#pragma unroll
  for (int i = 0; i < 8; ++i) { r[i] = lo[i]; r[i + 8] = hi[i]; }
  return r;
}

// Issue one K-stage of async copies: A 128x32 + Bt 128x32, 4 x b128 per thread.
__device__ __forceinline__ void stage_issue(const bf16_t* __restrict__ A, int lda,
                                            const bf16_t* __restrict__ Bt, int ldb,
                                            int rowBase, int colBase, int k0,
                                            bf16_t* As, bf16_t* Bs, int tid) {
#pragma unroll
  for (int i = 0; i < 2; ++i) {
    int c = tid + i * 256;            // chunk id in [0,512)
    int row = c >> 2, sub = c & 3;    // 4 x 8-elem chunks per row
    async_copy_b128(lds_off(As + row * LDP + sub * 8),
                    A + (size_t)(rowBase + row) * lda + k0 + sub * 8);
  }
#pragma unroll
  for (int i = 0; i < 2; ++i) {
    int c = tid + i * 256;
    int row = c >> 2, sub = c & 3;
    async_copy_b128(lds_off(Bs + row * LDP + sub * 8),
                    Bt + (size_t)(colBase + row) * ldb + k0 + sub * 8);
  }
}

// One K-stage of compute: 1 A fragment, 8 B fragments, 8 WMMAs back-to-back.
__device__ __forceinline__ void compute_stage(const bf16_t* As, const bf16_t* Bs,
                                              f32x8 acc[8], int wave, int r, int half) {
  bf16x16 a = lds_frag(As + wave * 16 * LDP, r, half);
  bf16x16 b[8];
#pragma unroll
  for (int j = 0; j < 8; ++j) b[j] = lds_frag(Bs + j * 16 * LDP, r, half);
#pragma unroll
  for (int j = 0; j < 8; ++j) acc[j] = wmma_bf16(a, b[j], acc[j]);
}

// ---------------------------------------------------------------- 1) fp32 -> bf16
__global__ void k_cvt_bf16(const float* __restrict__ in, bf16_t* __restrict__ out, int n4) {
  int i = blockIdx.x * blockDim.x + threadIdx.x;
  if (i >= n4) return;
  f32x4 v = *(const f32x4*)(in + (size_t)i * 4);
  bf16_t* o = out + (size_t)i * 4;
#pragma unroll
  for (int j = 0; j < 4; ++j) o[j] = (bf16_t)v[j];
}

// ---------------------------------------------------------------- 2) W [K][N] fp32 -> Wt [N][K] bf16
__global__ void k_transpose_w(const float* __restrict__ in, bf16_t* __restrict__ out) {
  __shared__ bf16_t tile[32][33];
  int bn = blockIdx.x * 32;   // N block
  int bk = blockIdx.y * 32;   // K block
  int tx = threadIdx.x, ty = threadIdx.y;      // (32, 8)
#pragma unroll
  for (int i = 0; i < 32; i += 8)
    tile[ty + i][tx] = (bf16_t)in[(size_t)(bk + ty + i) * DD + bn + tx];  // tile[k][n]
  __syncthreads();
#pragma unroll
  for (int i = 0; i < 32; i += 8)
    out[(size_t)(bn + ty + i) * DD + bk + tx] = tile[tx][ty + i];         // out[n][k]
}

// ---------------------------------------------------------------- 3) fused QKV projection
// Block-cooperative GEMM: macro tile 128(M) x 128(N), K-step 32, async double-buffered LDS,
// pipeline unrolled 2 stages so buffer indices are compile-time constants.
// grid.x = (8192/128)*(1024/128) = 512, grid.z = 0/1/2 (Q/K/V)
// Q,K -> [bh][s][64] bf16 ; V -> transposed [bh][d][s] bf16
__global__ void __launch_bounds__(256)
k_qkv(const bf16_t* __restrict__ xb,
      const bf16_t* __restrict__ WqT, const bf16_t* __restrict__ WkT,
      const bf16_t* __restrict__ WvT,
      const float* __restrict__ bq, const float* __restrict__ bk,
      const float* __restrict__ bv,
      bf16_t* __restrict__ q_ws, bf16_t* __restrict__ k_ws,
      bf16_t* __restrict__ vt_ws) {
  __shared__ __align__(16) bf16_t As[2][128 * LDP];
  __shared__ __align__(16) bf16_t Bs[2][128 * LDP];

  int z = blockIdx.z;
  const bf16_t* Bt  = (z == 0) ? WqT : (z == 1) ? WkT : WvT;
  const float* bias = (z == 0) ? bq  : (z == 1) ? bk  : bv;

  int tid = threadIdx.x;
  int wave = tid >> 5, lane = tid & 31;
  int r = lane & 15, half = lane >> 4;
  int rowBase = (blockIdx.x >> 3) * 128;
  int colBase = (blockIdx.x & 7) * 128;

  f32x8 acc[8] = {};

  const int NSTEPS = DD / 32;                 // 32 (even)
  stage_issue(xb, DD, Bt, DD, rowBase, colBase, 0, As[0], Bs[0], tid);
  for (int s = 0; s < NSTEPS; s += 2) {
    stage_issue(xb, DD, Bt, DD, rowBase, colBase, (s + 1) * 32, As[1], Bs[1], tid);
    wait_async_le(4);                         // stage s resident, stage s+1 in flight
    __syncthreads();
    compute_stage(As[0], Bs[0], acc, wave, r, half);
    __syncthreads();
    if (s + 2 < NSTEPS) {
      stage_issue(xb, DD, Bt, DD, rowBase, colBase, (s + 2) * 32, As[0], Bs[0], tid);
      wait_async_le(4);
    } else {
      wait_async_le(0);
    }
    __syncthreads();
    compute_stage(As[1], Bs[1], acc, wave, r, half);
    __syncthreads();
  }

#pragma unroll
  for (int j = 0; j < 8; ++j) {
    int n = colBase + j * 16 + r;
    int head = n >> 6, d = n & 63;
    float bs = bias[n];
#pragma unroll
    for (int v = 0; v < 8; ++v) {
      int row = rowBase + wave * 16 + v + 8 * half;   // global token index
      int b = row >> 10, sidx = row & 1023;
      int bh = b * HH + head;
      float val = acc[j][v] + bs;
      if (z == 0)      q_ws[((size_t)bh * SS + sidx) * HD + d] = (bf16_t)val;
      else if (z == 1) k_ws[((size_t)bh * SS + sidx) * HD + d] = (bf16_t)val;
      else             vt_ws[((size_t)bh * HD + d) * SS + sidx] = (bf16_t)val;
    }
  }
}

// ---------------------------------------------------------------- 4) scores + softmax -> attn (fp32, d_out)
// grid = (B*H, S/16), block = 256 (8 waves, each wave owns 128 keys)
__global__ void k_scores_softmax(const bf16_t* __restrict__ q_ws,
                                 const bf16_t* __restrict__ k_ws,
                                 const unsigned char* __restrict__ mask,
                                 float* __restrict__ attn) {
  __shared__ float red[8][16];
  int bh = blockIdx.x, b = bh >> 4;
  int qbase = blockIdx.y * 16;
  int wave = threadIdx.x >> 5, lane = threadIdx.x & 31;
  int r = lane & 15, half = lane >> 4;

  const bf16_t* Q = q_ws + (size_t)bh * SS * HD + (size_t)qbase * HD;
  const bf16_t* K = k_ws + (size_t)bh * SS * HD;

  bf16x16 a0 = load_frag(Q, r, HD, 0,  half);
  bf16x16 a1 = load_frag(Q, r, HD, 32, half);

  f32x8 sc[8];
#pragma unroll
  for (int t = 0; t < 8; ++t) {
    int colBase = wave * 128 + t * 16;
    bf16x16 b0 = load_frag(K + (size_t)colBase * HD, r, HD, 0,  half);
    bf16x16 b1 = load_frag(K + (size_t)colBase * HD, r, HD, 32, half);
    f32x8 c = {};
    c = wmma_bf16(a0, b0, c);
    c = wmma_bf16(a1, b1, c);
    bool ok = mask[(size_t)b * SS + colBase + r] != 0;
#pragma unroll
    for (int v = 0; v < 8; ++v) {
      float s = c[v] * 0.125f;                 // 1/sqrt(64)
      sc[t][v] = ok ? s : -1e9f;
    }
  }
  // per-wave row max (row = half*8+v lives across one 16-lane half)
  float rmax[8];
#pragma unroll
  for (int v = 0; v < 8; ++v) {
    float m = sc[0][v];
#pragma unroll
    for (int t = 1; t < 8; ++t) m = fmaxf(m, sc[t][v]);
#pragma unroll
    for (int sh = 1; sh < 16; sh <<= 1) m = fmaxf(m, __shfl_xor(m, sh));
    rmax[v] = m;
  }
  if (r == 0) {
#pragma unroll
    for (int v = 0; v < 8; ++v) red[wave][half * 8 + v] = rmax[v];
  }
  __syncthreads();
  float M[8];
#pragma unroll
  for (int v = 0; v < 8; ++v) {
    float m = -1e30f;
#pragma unroll
    for (int w = 0; w < 8; ++w) m = fmaxf(m, red[w][half * 8 + v]);
    M[v] = m;
  }
  __syncthreads();
  // exp + row sum
  float rsum[8];
#pragma unroll
  for (int v = 0; v < 8; ++v) rsum[v] = 0.f;
#pragma unroll
  for (int t = 0; t < 8; ++t)
#pragma unroll
    for (int v = 0; v < 8; ++v) {
      float e = __expf(sc[t][v] - M[v]);
      sc[t][v] = e;
      rsum[v] += e;
    }
#pragma unroll
  for (int v = 0; v < 8; ++v) {
    float s = rsum[v];
#pragma unroll
    for (int sh = 1; sh < 16; sh <<= 1) s += __shfl_xor(s, sh);
    rsum[v] = s;
  }
  if (r == 0) {
#pragma unroll
    for (int v = 0; v < 8; ++v) red[wave][half * 8 + v] = rsum[v];
  }
  __syncthreads();
  float inv[8];
#pragma unroll
  for (int v = 0; v < 8; ++v) {
    float s = 0.f;
#pragma unroll
    for (int w = 0; w < 8; ++w) s += red[w][half * 8 + v];
    inv[v] = 1.0f / s;
  }
  // write probabilities
  float* Arow = attn + ((size_t)bh * SS + qbase) * SS;
#pragma unroll
  for (int t = 0; t < 8; ++t) {
    int col = wave * 128 + t * 16 + r;
#pragma unroll
    for (int v = 0; v < 8; ++v) {
      int row = v + 8 * half;
      Arow[(size_t)row * SS + col] = sc[t][v] * inv[v];
    }
  }
}

// ---------------------------------------------------------------- 5) ctx = P @ V  -> ctx_ws [b][s][D_ATTN] bf16
// grid = (B*H, S/16), block = 128 (4 waves, wave w = d-tile w)
__global__ void k_ctx(const float* __restrict__ attn,
                      const bf16_t* __restrict__ vt_ws,
                      bf16_t* __restrict__ ctx_ws) {
  int bh = blockIdx.x, b = bh >> 4, head = bh & 15;
  int qbase = blockIdx.y * 16;
  int wave = threadIdx.x >> 5, lane = threadIdx.x & 31;
  int r = lane & 15, half = lane >> 4;

  const float*  P  = attn + ((size_t)bh * SS + qbase) * SS;
  const bf16_t* Vt = vt_ws + (size_t)bh * HD * SS + (size_t)(wave * 16) * SS;

  f32x8 acc = {};
  for (int kc = 0; kc < SS; kc += 32) {
    // A fragment: fp32 probabilities converted on load
    const float* p0 = P + (size_t)r * SS + kc + half * 8;
    __builtin_prefetch(p0 + 32, 0, 1);
    f32x4 x0 = *(const f32x4*)(p0);
    f32x4 x1 = *(const f32x4*)(p0 + 4);
    f32x4 y0 = *(const f32x4*)(p0 + 16);
    f32x4 y1 = *(const f32x4*)(p0 + 20);
    bf16x16 a;
#pragma unroll
    for (int i = 0; i < 4; ++i) {
      a[i]      = (bf16_t)x0[i];
      a[4 + i]  = (bf16_t)x1[i];
      a[8 + i]  = (bf16_t)y0[i];
      a[12 + i] = (bf16_t)y1[i];
    }
    bf16x16 bfr = load_frag(Vt, r, SS, kc, half);
    acc = wmma_bf16(a, bfr, acc);
  }
  int d = wave * 16 + r;
#pragma unroll
  for (int v = 0; v < 8; ++v) {
    int q = qbase + v + 8 * half;
    ctx_ws[((size_t)b * SS + q) * DD + head * HD + d] = (bf16_t)acc[v];
  }
}

// ---------------------------------------------------------------- 6) out = ctx @ Wo + bo + x -> pre (fp32)
// Same block-cooperative async-LDS GEMM as k_qkv (128x128 macro tile).
__global__ void __launch_bounds__(256)
k_outproj(const bf16_t* __restrict__ ctx_ws,
          const bf16_t* __restrict__ WoT,
          const float* __restrict__ bo,
          const float* __restrict__ x,
          float* __restrict__ pre) {
  __shared__ __align__(16) bf16_t As[2][128 * LDP];
  __shared__ __align__(16) bf16_t Bs[2][128 * LDP];

  int tid = threadIdx.x;
  int wave = tid >> 5, lane = tid & 31;
  int r = lane & 15, half = lane >> 4;
  int rowBase = (blockIdx.x >> 3) * 128;
  int colBase = (blockIdx.x & 7) * 128;

  f32x8 acc[8] = {};

  const int NSTEPS = DD / 32;
  stage_issue(ctx_ws, DD, WoT, DD, rowBase, colBase, 0, As[0], Bs[0], tid);
  for (int s = 0; s < NSTEPS; s += 2) {
    stage_issue(ctx_ws, DD, WoT, DD, rowBase, colBase, (s + 1) * 32, As[1], Bs[1], tid);
    wait_async_le(4);
    __syncthreads();
    compute_stage(As[0], Bs[0], acc, wave, r, half);
    __syncthreads();
    if (s + 2 < NSTEPS) {
      stage_issue(ctx_ws, DD, WoT, DD, rowBase, colBase, (s + 2) * 32, As[0], Bs[0], tid);
      wait_async_le(4);
    } else {
      wait_async_le(0);
    }
    __syncthreads();
    compute_stage(As[1], Bs[1], acc, wave, r, half);
    __syncthreads();
  }

#pragma unroll
  for (int j = 0; j < 8; ++j) {
    int n = colBase + j * 16 + r;
    float bs = bo[n];
#pragma unroll
    for (int v = 0; v < 8; ++v) {
      int row = rowBase + wave * 16 + v + 8 * half;
      pre[(size_t)row * DD + n] = acc[j][v] + bs + x[(size_t)row * DD + n];
    }
  }
}

// ---------------------------------------------------------------- 7) LayerNorm -> d_out
__global__ void k_ln(const float* __restrict__ pre,
                     const float* __restrict__ gamma, const float* __restrict__ beta,
                     float* __restrict__ out) {
  __shared__ float ls[8], ls2[8];
  int row = blockIdx.x;
  const float* y = pre + (size_t)row * DD;
  float s = 0.f, s2 = 0.f;
  for (int c = threadIdx.x; c < DD; c += 256) {
    float v = y[c];
    s += v; s2 += v * v;
  }
#pragma unroll
  for (int m = 1; m < 32; m <<= 1) { s += __shfl_xor(s, m); s2 += __shfl_xor(s2, m); }
  int wave = threadIdx.x >> 5, lane = threadIdx.x & 31;
  if (lane == 0) { ls[wave] = s; ls2[wave] = s2; }
  __syncthreads();
  s = 0.f; s2 = 0.f;
#pragma unroll
  for (int w = 0; w < 8; ++w) { s += ls[w]; s2 += ls2[w]; }
  float mu  = s * (1.0f / DD);
  float var = s2 * (1.0f / DD) - mu * mu;
  float inv = rsqrtf(var + 1e-5f);
  for (int c = threadIdx.x; c < DD; c += 256)
    out[(size_t)row * DD + c] = (y[c] - mu) * inv * gamma[c] + beta[c];
}

// ---------------------------------------------------------------- launcher
extern "C" void kernel_launch(void* const* d_in, const int* in_sizes, int n_in,
                              void* d_out, int out_size, void* d_ws, size_t ws_size,
                              hipStream_t stream) {
  const float* x            = (const float*)d_in[0];
  const unsigned char* mask = (const unsigned char*)d_in[1];
  const float* Wq = (const float*)d_in[2];  const float* bq = (const float*)d_in[3];
  const float* Wk = (const float*)d_in[4];  const float* bk = (const float*)d_in[5];
  const float* Wv = (const float*)d_in[6];  const float* bv = (const float*)d_in[7];
  const float* Wo = (const float*)d_in[8];  const float* bo = (const float*)d_in[9];
  const float* gamma = (const float*)d_in[10];
  const float* beta  = (const float*)d_in[11];

  float* out  = (float*)d_out;                          // [8,1024,1024]
  float* attn = out + (size_t)BB * SS * DD;             // [8,16,1024,1024]

  // workspace carve (256B aligned)
  char* ws = (char*)d_ws;
  size_t off = 0;
  auto take = [&](size_t bytes) { char* p = ws + off; off += (bytes + 255) & ~(size_t)255; return p; };
  bf16_t* xb     = (bf16_t*)take((size_t)BB * SS * DD * 2);
  bf16_t* WqT    = (bf16_t*)take((size_t)DD * DD * 2);
  bf16_t* WkT    = (bf16_t*)take((size_t)DD * DD * 2);
  bf16_t* WvT    = (bf16_t*)take((size_t)DD * DD * 2);
  bf16_t* WoT    = (bf16_t*)take((size_t)DD * DD * 2);
  bf16_t* q_wsp  = (bf16_t*)take((size_t)BB * HH * SS * HD * 2);
  bf16_t* k_wsp  = (bf16_t*)take((size_t)BB * HH * SS * HD * 2);
  bf16_t* vt_wsp = (bf16_t*)take((size_t)BB * HH * HD * SS * 2);
  bf16_t* ctx_ws = (bf16_t*)take((size_t)BB * SS * DD * 2);
  float*  pre_ws = (float*) take((size_t)BB * SS * DD * 4);

  // 1) x -> bf16
  int n4 = (BB * SS * DD) / 4;
  k_cvt_bf16<<<(n4 + 255) / 256, 256, 0, stream>>>(x, xb, n4);

  // 2) weight transpose+convert
  dim3 tb(32, 8), tg(DD / 32, DD / 32);
  k_transpose_w<<<tg, tb, 0, stream>>>(Wq, WqT);
  k_transpose_w<<<tg, tb, 0, stream>>>(Wk, WkT);
  k_transpose_w<<<tg, tb, 0, stream>>>(Wv, WvT);
  k_transpose_w<<<tg, tb, 0, stream>>>(Wo, WoT);

  // 3) QKV projection (async double-buffered LDS GEMM, 128x128 macro tile)
  k_qkv<<<dim3(512, 1, 3), 256, 0, stream>>>(xb, WqT, WkT, WvT, bq, bk, bv,
                                             q_wsp, k_wsp, vt_wsp);

  // 4) scores + softmax -> attn (d_out)
  k_scores_softmax<<<dim3(BB * HH, SS / 16), 256, 0, stream>>>(q_wsp, k_wsp, mask, attn);

  // 5) ctx = P @ V
  k_ctx<<<dim3(BB * HH, SS / 16), 128, 0, stream>>>(attn, vt_wsp, ctx_ws);

  // 6) out projection + residual (async double-buffered LDS GEMM, 128x128 macro tile)
  k_outproj<<<512, 256, 0, stream>>>(ctx_ws, WoT, bo, x, pre_ws);

  // 7) LayerNorm
  k_ln<<<BB * SS, 256, 0, stream>>>(pre_ws, gamma, beta, out);

  (void)in_sizes; (void)n_in; (void)out_size; (void)ws_size;
}
